// DiffusionTransformerBlock_84396107366690
// MI455X (gfx1250) — matrix operand order repaired and checked by reading, not compiled, not run on hardware
//
#include <hip/hip_runtime.h>
#include <hip/hip_bf16.h>

typedef unsigned short u16;
typedef __attribute__((ext_vector_type(16))) __bf16 bf16x16;
typedef __attribute__((ext_vector_type(8)))  float   f32x8;
typedef __attribute__((ext_vector_type(4)))  unsigned int u32x4;

// ---------- helpers ----------
__device__ __forceinline__ u16 f2bf(float x) {
    unsigned u = __float_as_uint(x);
    unsigned r = u + 0x7FFFu + ((u >> 16) & 1u);   // round-to-nearest-even
    return (u16)(r >> 16);
}

union Frag32B { struct { u32x4 lo, hi; } q; bf16x16 bf; };

__device__ __forceinline__ bf16x16 make_frag(u32x4 lo, u32x4 hi) {
    Frag32B f; f.q.lo = lo; f.q.hi = hi; return f.bf;
}

__device__ __forceinline__ float gelu_exact(float x) {
    return 0.5f * x * (1.0f + erff(x * 0.70710678118654752f));
}

// CDNA5 async global->LDS DMA (ASYNCcnt-tracked), GVS addressing:
// mem = SGPR_base + VGPR_i32_offset ; LDS dst = low 32 bits of flat shared ptr
__device__ __forceinline__ void async_copy_b128(unsigned lds_off, unsigned gl_off,
                                                const void* base) {
    asm volatile("global_load_async_to_lds_b128 %0, %1, %2"
                 :: "v"(lds_off), "v"(gl_off), "s"(base) : "memory");
}
__device__ __forceinline__ void wait_async0() {
    asm volatile("s_wait_asynccnt 0x0" ::: "memory");
}
__device__ __forceinline__ unsigned lds_addr(const void* p) {
    return (unsigned)(size_t)p;          // flat->LDS truncates to addr[31:0]
}

// ---------- tiny GEMM: t_emb @ wt + bt   (B x 2D outputs, B=2) ----------
__global__ void temb_kernel(const float* __restrict__ t_emb,
                            const float* __restrict__ wt,
                            const float* __restrict__ bt,
                            float* __restrict__ out, int D, int N2) {
    int i = blockIdx.x * blockDim.x + threadIdx.x;   // over B*N2
    int b = i / N2, j = i % N2;
    const float* te = t_emb + (size_t)b * D;
    float acc = bt[j];
    for (int d = 0; d < D; ++d) acc += te[d] * wt[(size_t)d * N2 + j];
    out[i] = acc;
}

// ---------- fp32 W[K][N] -> bf16 Wt[N][K] (transpose + convert) ----------
__global__ __launch_bounds__(256) void transcvt_kernel(const float* __restrict__ W,
                                                       u16* __restrict__ Wt,
                                                       int K, int N) {
    __shared__ float tile[32][33];
    int n0 = blockIdx.x * 32, k0 = blockIdx.y * 32;
    int tx = threadIdx.x, ty = threadIdx.y;          // 32 x 8
    for (int i = 0; i < 4; ++i)
        tile[ty + 8 * i][tx] = W[(size_t)(k0 + ty + 8 * i) * N + n0 + tx];
    __syncthreads();
    for (int i = 0; i < 4; ++i)
        Wt[(size_t)(n0 + ty + 8 * i) * K + k0 + tx] = f2bf(tile[tx][ty + 8 * i]);
}

// ---------- bf16 V[s][1024] (per head 64 cols) -> Vt[b][h][64][S] ----------
__global__ __launch_bounds__(256) void vtrans_kernel(const u16* __restrict__ v,
                                                     u16* __restrict__ vt, int S) {
    __shared__ u16 tile[32][34];
    int s0 = blockIdx.x * 32, d0 = blockIdx.y * 32;
    int bh = blockIdx.z;                              // b*16 + head
    int b = bh >> 4, head = bh & 15;
    int tx = threadIdx.x, ty = threadIdx.y;
    const u16* vb = v + (size_t)b * S * 1024 + head * 64;
    for (int i = 0; i < 4; ++i)
        tile[ty + 8 * i][tx] = vb[(size_t)(s0 + ty + 8 * i) * 1024 + d0 + tx];
    __syncthreads();
    u16* ob = vt + ((size_t)bh * 64) * S;
    for (int i = 0; i < 4; ++i)
        ob[(size_t)(d0 + ty + 8 * i) * S + s0 + tx] = tile[tx][ty + 8 * i];
}

// ---------- fused LayerNorm + adaLN modulate -> bf16 ----------
__global__ __launch_bounds__(256) void ln_mod_kernel(const float* __restrict__ x,
                                                     const float* __restrict__ w,
                                                     const float* __restrict__ bsh,
                                                     const float* __restrict__ t, // [B][2D]: shift | scale
                                                     u16* __restrict__ out,
                                                     int S, int D) {
    int row = blockIdx.x;                // 0..B*S-1
    int bidx = row / S;
    __shared__ float red1[256], red2[256];
    const float* xr = x + (size_t)row * D;
    float s1 = 0.f, s2 = 0.f;
    for (int i = threadIdx.x; i < D; i += 256) { float v = xr[i]; s1 += v; s2 += v * v; }
    red1[threadIdx.x] = s1; red2[threadIdx.x] = s2;
    __syncthreads();
    for (int off = 128; off > 0; off >>= 1) {
        if (threadIdx.x < off) {
            red1[threadIdx.x] += red1[threadIdx.x + off];
            red2[threadIdx.x] += red2[threadIdx.x + off];
        }
        __syncthreads();
    }
    float mean = red1[0] / (float)D;
    float var  = red2[0] / (float)D - mean * mean;
    float rstd = rsqrtf(var + 1e-5f);
    const float* shiftp = t + (size_t)bidx * (2 * D);
    const float* scalep = shiftp + D;
    u16* orow = out + (size_t)row * D;
    for (int i = threadIdx.x; i < D; i += 256) {
        float xn = (xr[i] - mean) * rstd * w[i] + bsh[i];
        orow[i] = f2bf(xn * (1.f + scalep[i]) + shiftp[i]);
    }
}

// ---------- WMMA GEMM: C = A[MxK](bf16) x Bt[NxK]^T(bf16) + bias ----------
// block = 256 thr (8 waves as 4x2), block tile 128x128, wave tile 32x64
// Double-buffered LDS; tiles staged with GLOBAL_LOAD_ASYNC_TO_LDS_B128 (ASYNCcnt),
// software-pipelined: issue async for tile t+1, run WMMA on tile t, wait+barrier.
// EPI: 0 = bias -> bf16 out ; 1 = bias+GELU -> bf16 out ; 2 = bias+residual -> f32 out
template <int EPI>
__global__ __launch_bounds__(256) void gemm_bf16_kernel(
    const u16* __restrict__ A, const u16* __restrict__ Bt,
    const float* __restrict__ bias, const float* __restrict__ res,
    float* __restrict__ outF, u16* __restrict__ outH,
    int M, int N, int K) {
    __shared__ u16 As[2][128 * 40];   // [m][k], rows padded to 40 -> conflict-free frags
    __shared__ u16 Bs[2][128 * 40];   // [n][k]
    const int tid  = threadIdx.x;
    const int lane = tid & 31;
    const int wave = tid >> 5;
    const int wm = wave >> 1, wn = wave & 1;
    const int m0 = blockIdx.y * 128, n0 = blockIdx.x * 128;
    const int ln = lane & 15, lh = lane >> 4;

    // this thread's two 16B chunks of each tile
    const int rr0 = tid >> 2,          cc0 = (tid & 3) * 8;
    const int rr1 = (tid + 256) >> 2,  cc1 = ((tid + 256) & 3) * 8;

    auto issue_async = [&](int bi, int kt) {
        async_copy_b128(lds_addr(&As[bi][rr0 * 40 + cc0]),
                        (unsigned)(((size_t)(m0 + rr0) * K + kt + cc0) * 2), A);
        async_copy_b128(lds_addr(&Bs[bi][rr0 * 40 + cc0]),
                        (unsigned)(((size_t)(n0 + rr0) * K + kt + cc0) * 2), Bt);
        async_copy_b128(lds_addr(&As[bi][rr1 * 40 + cc1]),
                        (unsigned)(((size_t)(m0 + rr1) * K + kt + cc1) * 2), A);
        async_copy_b128(lds_addr(&Bs[bi][rr1 * 40 + cc1]),
                        (unsigned)(((size_t)(n0 + rr1) * K + kt + cc1) * 2), Bt);
    };

    f32x8 acc[2][4];
    for (int i = 0; i < 2; ++i)
        for (int j = 0; j < 4; ++j)
            for (int r = 0; r < 8; ++r) acc[i][j][r] = 0.f;

    const int NT = K / 32;
    issue_async(0, 0);
    wait_async0();
    __syncthreads();

    for (int t = 0; t < NT; ++t) {
        const int cur = t & 1;
        if (t + 1 < NT) issue_async(1 - cur, (t + 1) * 32);   // overlap DMA with WMMA

        bf16x16 af[2], bfm[4];
        for (int i = 0; i < 2; ++i) {
            const u16* ap = &As[cur][(wm * 32 + i * 16 + ln) * 40];
            af[i] = make_frag(*(const u32x4*)(ap + lh * 8),
                              *(const u32x4*)(ap + 16 + lh * 8));
        }
        for (int j = 0; j < 4; ++j) {
            const u16* bp = &Bs[cur][(wn * 64 + j * 16 + ln) * 40];
            bfm[j] = make_frag(*(const u32x4*)(bp + lh * 8),
                               *(const u32x4*)(bp + 16 + lh * 8));
        }
        for (int i = 0; i < 2; ++i)
            for (int j = 0; j < 4; ++j)
                acc[i][j] = __builtin_amdgcn_wmma_f32_16x16x32_bf16(
                    false, af[i], false, bfm[j], (short)0, acc[i][j], false, false);

        wait_async0();        // own DMAs done (writes to other buffer)
        __syncthreads();      // everyone done reading cur + writing next
    }

    // epilogue (C layout: lane holds col n=ln, rows r + 8*lh)
    for (int j = 0; j < 4; ++j) {
        int n = n0 + wn * 64 + j * 16 + ln;
        float bv = bias[n];
        for (int i = 0; i < 2; ++i) {
            for (int r = 0; r < 8; ++r) {
                int m = m0 + wm * 32 + i * 16 + r + 8 * lh;
                size_t o = (size_t)m * N + n;
                float v = acc[i][j][r] + bv;
                if (EPI == 1) v = gelu_exact(v);
                if (EPI == 2) outF[o] = v + res[o];
                else          outH[o] = f2bf(v);
            }
        }
    }
}

// ---------- flash attention (bf16 WMMA, fp32 online softmax) ----------
// grid: (S/128, H, B); block: 256 (8 waves, 16 query rows each)
__global__ __launch_bounds__(256) void attn_kernel(const u16* __restrict__ q,
                                                   const u16* __restrict__ k,
                                                   const u16* __restrict__ vt,
                                                   u16* __restrict__ attn_out, int S) {
    __shared__ u16 Ps[8][16 * 72];     // per-wave P staging, row pad 72 (144B, 16B-aligned)
    const int tid = threadIdx.x, lane = tid & 31, wave = tid >> 5;
    const int ln = lane & 15, lh = lane >> 4;
    const int head = blockIdx.y, b = blockIdx.z;
    const int q0 = blockIdx.x * 128 + wave * 16;
    const float scale = 0.125f;        // 1/sqrt(64)

    // Q A-fragments (rows fixed for whole kernel)
    bf16x16 aq[2];
    {
        const u16* qp = q + ((size_t)(b * S) + q0 + ln) * 1024 + head * 64;
        for (int ks = 0; ks < 2; ++ks)
            aq[ks] = make_frag(*(const u32x4*)(qp + ks * 32 + lh * 8),
                               *(const u32x4*)(qp + ks * 32 + 16 + lh * 8));
    }
    const u16* kbh = k + (size_t)b * S * 1024 + head * 64;
    const u16* vtb = vt + ((size_t)(b * 16 + head) * 64) * S;   // [64][S]

    float mrun[8], lrun[8];
    f32x8 o[4];
    for (int r = 0; r < 8; ++r) { mrun[r] = -3.0e38f; lrun[r] = 0.f; }
    for (int j = 0; j < 4; ++j) for (int r = 0; r < 8; ++r) o[j][r] = 0.f;
    u16* pws = &Ps[wave][0];

    for (int key0 = 0; key0 < S; key0 += 64) {
        // S_tile = Q @ K^T  (K rows are contiguous in hd == WMMA-K)
        f32x8 s4[4];
        for (int j = 0; j < 4; ++j) for (int r = 0; r < 8; ++r) s4[j][r] = 0.f;
        for (int ks = 0; ks < 2; ++ks)
            for (int j = 0; j < 4; ++j) {
                const u16* kp = kbh + (size_t)(key0 + j * 16 + ln) * 1024 + ks * 32;
                bf16x16 bk = make_frag(*(const u32x4*)(kp + lh * 8), *(const u32x4*)(kp + 16 + lh * 8));
                s4[j] = __builtin_amdgcn_wmma_f32_16x16x32_bf16(
                    false, aq[ks], false, bk, (short)0, s4[j], false, false);
            }
        if (key0 + 64 < S) {   // global_prefetch_b8 on next K/V tiles
            __builtin_prefetch(kbh + (size_t)(key0 + 64 + ln) * 1024, 0, 3);
            __builtin_prefetch(vtb + (size_t)ln * S + key0 + 64, 0, 3);
        }
        // online softmax: row max over 4 n-tiles + 16-lane butterfly (rows differ per half)
        float nm[8];
        for (int r = 0; r < 8; ++r) {
            float mx = s4[0][r];
            for (int j = 1; j < 4; ++j) mx = fmaxf(mx, s4[j][r]);
            mx = fmaxf(mx, __shfl_xor(mx, 1, 32));
            mx = fmaxf(mx, __shfl_xor(mx, 2, 32));
            mx = fmaxf(mx, __shfl_xor(mx, 4, 32));
            mx = fmaxf(mx, __shfl_xor(mx, 8, 32));
            mx *= scale;
            nm[r] = fmaxf(mrun[r], mx);
            float corr = __expf(mrun[r] - nm[r]);
            lrun[r] *= corr;
            for (int j = 0; j < 4; ++j) o[j][r] *= corr;
            mrun[r] = nm[r];
        }
        // P = exp(scale*S - m); stage to LDS in A-matrix [m][k] order; accumulate row sums
        for (int r = 0; r < 8; ++r) {
            float rs = 0.f;
            for (int j = 0; j < 4; ++j) {
                float p = __expf(s4[j][r] * scale - nm[r]);
                rs += p;
                pws[(r + 8 * lh) * 72 + j * 16 + ln] = f2bf(p);
            }
            rs += __shfl_xor(rs, 1, 32);
            rs += __shfl_xor(rs, 2, 32);
            rs += __shfl_xor(rs, 4, 32);
            rs += __shfl_xor(rs, 8, 32);
            lrun[r] += rs;
        }
        // O += P @ V  (Vt rows are contiguous in s == WMMA-K); LDS in-order per wave
        for (int ks = 0; ks < 2; ++ks) {
            const u16* pp = pws + (size_t)ln * 72 + ks * 32;
            bf16x16 ap = make_frag(*(const u32x4*)(pp + lh * 8),
                                   *(const u32x4*)(pp + 16 + lh * 8));
            for (int j = 0; j < 4; ++j) {
                const u16* vp = vtb + (size_t)(j * 16 + ln) * S + key0 + ks * 32;
                bf16x16 bv = make_frag(*(const u32x4*)(vp + lh * 8), *(const u32x4*)(vp + 16 + lh * 8));
                o[j] = __builtin_amdgcn_wmma_f32_16x16x32_bf16(
                    false, ap, false, bv, (short)0, o[j], false, false);
            }
        }
    }
    // normalize + write bf16
    u16* ob = attn_out + ((size_t)(b * S) + q0) * 1024 + head * 64;
    for (int j = 0; j < 4; ++j)
        for (int r = 0; r < 8; ++r)
            ob[(size_t)(r + 8 * lh) * 1024 + j * 16 + ln] = f2bf(o[j][r] / lrun[r]);
}

// =================== launcher ===================
extern "C" void kernel_launch(void* const* d_in, const int* in_sizes, int n_in,
                              void* d_out, int out_size, void* d_ws, size_t ws_size,
                              hipStream_t stream) {
    const float* x       = (const float*)d_in[0];
    const float* t_emb   = (const float*)d_in[1];
    const float* ln1_w   = (const float*)d_in[2];
    const float* ln1_b   = (const float*)d_in[3];
    const float* ln2_w   = (const float*)d_in[4];
    const float* ln2_b   = (const float*)d_in[5];
    const float* wq = (const float*)d_in[6];  const float* bq = (const float*)d_in[7];
    const float* wk = (const float*)d_in[8];  const float* bk = (const float*)d_in[9];
    const float* wv = (const float*)d_in[10]; const float* bv = (const float*)d_in[11];
    const float* wo = (const float*)d_in[12]; const float* bo = (const float*)d_in[13];
    const float* wt_attn = (const float*)d_in[14]; const float* bt_attn = (const float*)d_in[15];
    const float* wt_ffn  = (const float*)d_in[16]; const float* bt_ffn  = (const float*)d_in[17];
    const float* w1 = (const float*)d_in[18]; const float* b1 = (const float*)d_in[19];
    const float* w2 = (const float*)d_in[20]; const float* b2 = (const float*)d_in[21];

    const int B = 2, S = 2048, D = 1024, F = 4096, M = B * S, H = 16;

    char* ws = (char*)d_ws;
    size_t off = 0;
    auto alloc = [&](size_t bytes) -> char* {
        char* p = ws + off;
        off += (bytes + 255) & ~(size_t)255;
        return p;
    };
    float* t_attn  = (float*)alloc((size_t)B * 2 * D * 4);
    float* t_ffn   = (float*)alloc((size_t)B * 2 * D * 4);
    u16*  xn       = (u16*)alloc((size_t)M * D * 2);       // reused for FFN input
    u16*  wqT      = (u16*)alloc((size_t)D * D * 2);
    u16*  wkT      = (u16*)alloc((size_t)D * D * 2);
    u16*  wvT      = (u16*)alloc((size_t)D * D * 2);
    u16*  woT      = (u16*)alloc((size_t)D * D * 2);
    u16*  w1T      = (u16*)alloc((size_t)F * D * 2);       // [F][D]
    u16*  w2T      = (u16*)alloc((size_t)D * F * 2);       // [D][F]
    u16*  qb       = (u16*)alloc((size_t)M * D * 2);
    u16*  kb       = (u16*)alloc((size_t)M * D * 2);
    u16*  vb       = (u16*)alloc((size_t)M * D * 2);
    u16*  vtb      = (u16*)alloc((size_t)M * D * 2);       // [B][H][64][S]
    u16*  attn_o   = (u16*)alloc((size_t)M * D * 2);
    float* x_attn  = (float*)alloc((size_t)M * D * 4);
    u16*  hbuf     = (u16*)alloc((size_t)M * F * 2);
    (void)ws_size; (void)in_sizes; (void)n_in; (void)out_size;

    dim3 b256(256), tb(32, 8);

    // 1) timestep modulation vectors
    temb_kernel<<<dim3((B * 2 * D) / 256), b256, 0, stream>>>(t_emb, wt_attn, bt_attn, t_attn, D, 2 * D);
    temb_kernel<<<dim3((B * 2 * D) / 256), b256, 0, stream>>>(t_emb, wt_ffn,  bt_ffn,  t_ffn,  D, 2 * D);

    // 2) transposed bf16 weights
    transcvt_kernel<<<dim3(D / 32, D / 32), tb, 0, stream>>>(wq, wqT, D, D);
    transcvt_kernel<<<dim3(D / 32, D / 32), tb, 0, stream>>>(wk, wkT, D, D);
    transcvt_kernel<<<dim3(D / 32, D / 32), tb, 0, stream>>>(wv, wvT, D, D);
    transcvt_kernel<<<dim3(D / 32, D / 32), tb, 0, stream>>>(wo, woT, D, D);
    transcvt_kernel<<<dim3(F / 32, D / 32), tb, 0, stream>>>(w1, w1T, D, F);
    transcvt_kernel<<<dim3(D / 32, F / 32), tb, 0, stream>>>(w2, w2T, F, D);

    // 3) LN1 + modulate
    ln_mod_kernel<<<dim3(M), b256, 0, stream>>>(x, ln1_w, ln1_b, t_attn, xn, S, D);

    // 4) QKV projections (WMMA + async-LDS pipeline)
    gemm_bf16_kernel<0><<<dim3(D / 128, M / 128), b256, 0, stream>>>(xn, wqT, bq, nullptr, nullptr, qb, M, D, D);
    gemm_bf16_kernel<0><<<dim3(D / 128, M / 128), b256, 0, stream>>>(xn, wkT, bk, nullptr, nullptr, kb, M, D, D);
    gemm_bf16_kernel<0><<<dim3(D / 128, M / 128), b256, 0, stream>>>(xn, wvT, bv, nullptr, nullptr, vb, M, D, D);

    // 5) V transpose per head, then flash attention (WMMA)
    vtrans_kernel<<<dim3(S / 32, 2, B * H), tb, 0, stream>>>(vb, vtb, S);
    attn_kernel<<<dim3(S / 128, H, B), b256, 0, stream>>>(qb, kb, vtb, attn_o, S);

    // 6) O projection + residual -> x_attn (fp32)
    gemm_bf16_kernel<2><<<dim3(D / 128, M / 128), b256, 0, stream>>>(attn_o, woT, bo, x, x_attn, nullptr, M, D, D);

    // 7) LN2 + modulate (reuse xn buffer)
    ln_mod_kernel<<<dim3(M), b256, 0, stream>>>(x_attn, ln2_w, ln2_b, t_ffn, xn, S, D);

    // 8) FFN: GEMM+GELU, then GEMM+residual -> d_out (fp32)
    gemm_bf16_kernel<1><<<dim3(F / 128, M / 128), b256, 0, stream>>>(xn, w1T, b1, nullptr, nullptr, hbuf, M, F, D);
    gemm_bf16_kernel<2><<<dim3(D / 128, M / 128), b256, 0, stream>>>(hbuf, w2T, b2, x_attn, (float*)d_out, nullptr, M, D, F);
}
